// Attention_2104533975550
// MI455X (gfx1250) — compile-verified, matrix operand order checked by reading, not run on hardware
//
#include <hip/hip_runtime.h>
#include <hip/hip_bf16.h>

#define B_   32
#define S_   2048
#define ENC_ 1024
#define DEC_ 1024

typedef __attribute__((ext_vector_type(16))) __bf16 v16bf;
typedef __attribute__((ext_vector_type(8)))  float  v8f;

// ---------------- helpers ----------------

__device__ __forceinline__ unsigned short f32_to_bf16(float f) {
    unsigned int u = __float_as_uint(f);
    u += 0x7FFFu + ((u >> 16) & 1u);        // round-to-nearest-even
    return (unsigned short)(u >> 16);
}

// tanh via v_exp_f32 (TRANS pipe), overflow-safe
__device__ __forceinline__ float fast_tanh(float x) {
    float ax = fabsf(x);
    float e  = __builtin_amdgcn_exp2f(ax * 2.88539008177792681f);  // exp2(2x*log2e)
    float r  = 1.0f - 2.0f / (e + 1.0f);
    return copysignf(r, x);
}

// Generic LDS pointer -> 32-bit LDS byte address (low 32 bits of shared aperture addr)
__device__ __forceinline__ unsigned lds_addr_of(const void* p) {
    return (unsigned)(unsigned long long)p;
}

// Issue an async 16KB memory->LDS copy: 256 threads x 4 x B128.
// GVS addressing: mem = SADDR + VADDR + IOFFSET ; LDS = VDST + IOFFSET  (IOFFSET hits both)
__device__ __forceinline__ void async_copy_16k(const unsigned short* __restrict__ gsrc,
                                               unsigned lds_dst_base, int tid) {
    unsigned lds_addr = lds_dst_base + (unsigned)(tid << 4);
    unsigned voff     = (unsigned)(tid << 4);
    asm volatile(
        "global_load_async_to_lds_b128 %0, %1, %2 offset:0\n\t"
        "global_load_async_to_lds_b128 %0, %1, %2 offset:4096\n\t"
        "global_load_async_to_lds_b128 %0, %1, %2 offset:8192\n\t"
        "global_load_async_to_lds_b128 %0, %1, %2 offset:12288"
        :: "v"(lds_addr), "v"(voff), "s"(gsrc)
        : "memory");
}

__device__ __forceinline__ void wait_async0() {
    asm volatile("s_wait_asynccnt 0x0" ::: "memory");
}

// ---------------- kernel 1: W_e -> pre-swizzled bf16 B-fragments ----------------
// B-fragment (32x16 bf16, K x N) per-lane layout (mirrors documented A layout):
//   lane = 16*h + n  holds column N=n ; j=0..7 -> K = 8h+j ; j=8..15 -> K = 16+8h+(j-8)
// Workspace order: [ntile(64)][echunk(32)][lane(32)][j(16)]  => 2 MiB
__global__ void k_convertB(const float* __restrict__ Wattn, unsigned short* __restrict__ wsB) {
    int idx = blockIdx.x * 256 + threadIdx.x;          // < 1<<20
    int j    =  idx        & 15;
    int lane = (idx >> 4)  & 31;
    int ec   = (idx >> 9)  & 31;
    int nt   =  idx >> 14;
    int n  = lane & 15;
    int cc = lane >> 4;
    int K  = (j & 7) + (cc << 3) + ((j & 8) ? 16 : 0);
    int e  = (ec << 5) + K;
    int k  = (nt << 4) + n;
    wsB[idx] = f32_to_bf16(Wattn[(size_t)k * (ENC_ + DEC_) + DEC_ + e]);
}

// ---------------- kernel 2: proj_h[b,k] = sum_d hidden[b,d] * W_attn[k,d] ----------------
__global__ void k_proj_h(const float* __restrict__ hidden, const float* __restrict__ Wattn,
                         float* __restrict__ projH) {
    __shared__ float sh[DEC_];
    int b = blockIdx.y;
    int k = blockIdx.x * 256 + threadIdx.x;
    for (int i = threadIdx.x; i < DEC_; i += 256) sh[i] = hidden[(size_t)b * DEC_ + i];
    __syncthreads();
    const float* wr = Wattn + (size_t)k * (ENC_ + DEC_);
    float acc = 0.f;
#pragma unroll 4
    for (int d = 0; d < DEC_; ++d) acc = fmaf(sh[d], wr[d], acc);
    projH[(size_t)b * DEC_ + k] = acc;
}

// ---------------- kernel 3: fused energy GEMM + tanh + W_v reduction -> scores ----------------
// grid (S/128, B), 256 threads (8 waves). Wave w owns s-rows [s0+16w, s0+16w+16).
// LDS: A frags 256KB | B double-buffer 2x16KB | proj_h 4KB | W_v 4KB  = 296KB
// 128 pipeline stages (64 ntiles x 2 halves of 16 echunks); stage s+1's B tile streams
// into LDS via global_load_async_to_lds_b128 while stage s runs its 16 WMMAs.
__global__ void k_scores(const float* __restrict__ enc,
                         const unsigned short* __restrict__ wsB,
                         const float* __restrict__ projH,
                         const float* __restrict__ Wv,
                         float* __restrict__ wsScores) {
    extern __shared__ unsigned char smem_raw[];
    unsigned short* sA  = (unsigned short*)smem_raw;                       // 262144 B
    unsigned short* sB  = (unsigned short*)(smem_raw + 262144);            // 2 x 16384 B
    float*          sPh = (float*)(smem_raw + 262144 + 32768);             // 1024
    float*          sWv = sPh + 1024;                                      // 1024

    const int tid  = threadIdx.x;
    const int w    = tid >> 5;      // wave id = mtile
    const int lane = tid & 31;
    const int b    = blockIdx.y;
    const int s0   = blockIdx.x * 128;

    // --- stage proj_h row and W_v ---
    for (int i = tid; i < 1024; i += 256) { sPh[i] = projH[(size_t)b * DEC_ + i]; sWv[i] = Wv[i]; }

    // --- prologue: kick off async fetch of B stage 0 ---
    const unsigned ldsB = lds_addr_of(sB);
    async_copy_16k(wsB, ldsB, tid);

    // --- stage A: enc[b, s0:s0+128, :] -> bf16, WMMA A-fragment swizzle ---
    // A-frag layout (16x32 bf16, MxK): lane = 16c+m ; j=0..7 -> K=8c+j ; j=8..15 -> K=16+8c+(j-8)
    const float4* src = (const float4*)(enc + ((size_t)b * S_ + s0) * ENC_);
    for (int i = tid; i < 128 * 256; i += 256) {       // 128 rows * 256 float4 per row
        int row = i >> 8;
        int e4  = (i & 255) << 2;                      // multiple of 4: never straddles a j-oct
        float4 v = src[i];
        int mtile = row >> 4, m = row & 15;
        int ec = e4 >> 5;
        int K0 = e4 & 31;
        int cc = (K0 >> 3) & 1;
        int jb = (K0 & 7) + ((K0 & 16) ? 8 : 0);
        unsigned short* dst = &sA[(((mtile << 5) + ec) << 9) + ((cc << 4 | m) << 4) + jb];
        dst[0] = f32_to_bf16(v.x); dst[1] = f32_to_bf16(v.y);
        dst[2] = f32_to_bf16(v.z); dst[3] = f32_to_bf16(v.w);
    }

    float part[8];
#pragma unroll
    for (int r = 0; r < 8; ++r) part[r] = 0.f;

    const unsigned short* aWave = &sA[((w << 5) << 9) + (lane << 4)];
    v8f c = {0.f, 0.f, 0.f, 0.f, 0.f, 0.f, 0.f, 0.f};

    for (int stage = 0; stage < 128; ++stage) {
        const int buf  = stage & 1;            // buffer holding this stage's B tile
        const int half = stage & 1;            // which 16-echunk half of the ntile

        wait_async0();                         // my async issues are done in LDS
        __syncthreads();                       // everyone's done + prev reads of buf^1 finished

        if (stage + 1 < 128)                   // uniform branch (EXEC stays all-ones)
            async_copy_16k(wsB + (size_t)(stage + 1) * 8192,
                           ldsB + (unsigned)(((stage + 1) & 1) << 14), tid);

        const unsigned short* bBuf   = sB + (buf << 13) + (lane << 4);
        const unsigned short* aStage = aWave + ((half << 4) << 9);
#pragma unroll 4
        for (int ec = 0; ec < 16; ++ec) {
            v16bf a  = *(const v16bf*)(aStage + (ec << 9));
            v16bf bb = *(const v16bf*)(bBuf + (ec << 9));
            c = __builtin_amdgcn_wmma_f32_16x16x32_bf16(
                    false, a, false, bb, (short)0, c, false, false);
        }

        if (half == 1) {                       // ntile complete: fold tanh + W_v, reset C
            int nt = stage >> 1;
            // C/D layout: lane(0-15)->N=lane M=r ; lane(16-31)->N=lane-16 M=r+8
            int n  = lane & 15;
            int k  = (nt << 4) + n;
            float ph = sPh[k];
            float wv = sWv[k];
#pragma unroll
            for (int r = 0; r < 8; ++r) {
                part[r] = fmaf(wv, fast_tanh(c[r] + ph), part[r]);
                c[r] = 0.f;
            }
        }
    }

    // reduce over N (16 lanes within each half) via butterfly
#pragma unroll
    for (int r = 0; r < 8; ++r) {
        float v = part[r];
        v += __shfl_xor(v, 1, 32);
        v += __shfl_xor(v, 2, 32);
        v += __shfl_xor(v, 4, 32);
        v += __shfl_xor(v, 8, 32);
        part[r] = v;
    }
    if ((lane & 15) == 0) {
        int h = lane >> 4;
#pragma unroll
        for (int r = 0; r < 8; ++r) {
            int srow = s0 + (w << 4) + (h << 3) + r;
            wsScores[(size_t)b * S_ + srow] = part[r];
        }
    }
}

// ---------------- kernel 4: masked softmax over S per batch row ----------------
__global__ void k_softmax(const float* __restrict__ wsScores, const int* __restrict__ mask,
                          float* __restrict__ outW) {
    __shared__ float red[256];
    int b = blockIdx.x, tid = threadIdx.x;
    float vals[8];
    float mx = -3.4e38f;
#pragma unroll
    for (int i = 0; i < 8; ++i) {
        int s = tid + i * 256;
        float v = wsScores[(size_t)b * S_ + s];
        if (mask[(size_t)b * S_ + s] == 0) v = -1.0e9f;
        vals[i] = v;
        mx = fmaxf(mx, v);
    }
    red[tid] = mx; __syncthreads();
    for (int off = 128; off > 0; off >>= 1) {
        if (tid < off) red[tid] = fmaxf(red[tid], red[tid + off]);
        __syncthreads();
    }
    mx = red[0]; __syncthreads();

    float sum = 0.f;
#pragma unroll
    for (int i = 0; i < 8; ++i) {
        vals[i] = __builtin_amdgcn_exp2f((vals[i] - mx) * 1.44269504088896340f);
        sum += vals[i];
    }
    red[tid] = sum; __syncthreads();
    for (int off = 128; off > 0; off >>= 1) {
        if (tid < off) red[tid] += red[tid + off];
        __syncthreads();
    }
    float inv = 1.0f / red[0];
#pragma unroll
    for (int i = 0; i < 8; ++i)
        outW[(size_t)b * S_ + tid + i * 256] = vals[i] * inv;
}

// ---------------- kernel 5: context[b,e] = sum_s weights[b,s]*enc[b,s,e] ----------------
__global__ void k_context(const float* __restrict__ enc, const float* __restrict__ wts,
                          float* __restrict__ outC) {
    __shared__ float sw[S_];
    int b = blockIdx.y;
    int e = blockIdx.x * 256 + threadIdx.x;
    for (int i = threadIdx.x; i < S_; i += 256) sw[i] = wts[(size_t)b * S_ + i];
    __syncthreads();
    const float* ep = enc + (size_t)b * S_ * ENC_ + e;
    float acc = 0.f;
#pragma unroll 4
    for (int s = 0; s < S_; ++s) acc = fmaf(sw[s], ep[(size_t)s * ENC_], acc);
    outC[(size_t)b * ENC_ + e] = acc;
}

// ---------------- launcher ----------------
extern "C" void kernel_launch(void* const* d_in, const int* in_sizes, int n_in,
                              void* d_out, int out_size, void* d_ws, size_t ws_size,
                              hipStream_t stream) {
    const float* hidden = (const float*)d_in[0];   // (32,1024)
    const float* enc    = (const float*)d_in[1];   // (32,2048,1024)
    const int*   mask   = (const int*)  d_in[2];   // (32,2048)
    const float* Wattn  = (const float*)d_in[3];   // (1024,2048)
    const float* Wv     = (const float*)d_in[4];   // (1,1024)

    float* outC = (float*)d_out;                   // context: 32*1024
    float* outW = outC + (size_t)B_ * ENC_;        // weights: 32*2048

    unsigned short* wsB      = (unsigned short*)d_ws;                              // 2 MiB
    float*          wsProjH  = (float*)((char*)d_ws + 2u * 1024 * 1024);           // 128 KiB
    float*          wsScores = (float*)((char*)d_ws + 2u * 1024 * 1024 + 131072);  // 256 KiB

    k_convertB<<<4096, 256, 0, stream>>>(Wattn, wsB);
    k_proj_h <<<dim3(4, 32), 256, 0, stream>>>(hidden, Wattn, wsProjH);

    const size_t smem = 262144 + 32768 + 4096 + 4096; // 296 KiB (<= 320 KiB WGP LDS)
    k_scores  <<<dim3(16, 32), 256, smem, stream>>>(enc, wsB, wsProjH, Wv, wsScores);

    k_softmax <<<32, 256, 0, stream>>>(wsScores, mask, outW);
    k_context <<<dim3(4, 32), 256, 0, stream>>>(enc, outW, outC);
}
